// MyModel_87522843558713
// MI455X (gfx1250) — matrix operand order repaired and checked by reading, not compiled, not run on hardware
//
#include <hip/hip_runtime.h>
#include <hip/hip_bf16.h>
#include <math.h>
#include <stdint.h>

// Problem dims
#define Bb 128
#define Ss 256
#define Ee 128
#define Uu 512
#define Cc 3
#define NT (Uu / 16)   // 32 hidden-unit tiles
#define MT (Bb / 16)   // 8 batch tiles (one wave each)

typedef __attribute__((ext_vector_type(16))) _Float16 v16h;
typedef __attribute__((ext_vector_type(8)))  _Float16 v8h;
typedef __attribute__((ext_vector_type(8)))  float    v8f;
typedef __attribute__((ext_vector_type(4)))  int      v4i;
typedef __attribute__((address_space(1)))    v4i      gas_v4i;  // global
typedef __attribute__((address_space(3)))    v4i      las_v4i;  // LDS

#if defined(__HIP_DEVICE_COMPILE__) && __has_builtin(__builtin_amdgcn_global_load_async_to_lds_b128)
#define USE_ASYNC_LDS 1
#else
#define USE_ASYNC_LDS 0
#endif

struct DirArgs {
  const _Float16* xseq;    // [S, B, Kin] layer input sequence (f16)
  const _Float16* pw;      // packed weights [(g*NT+nt)*KT + kt][lane 0..31][16 halves]
  const float*    bias;    // [4U]
  _Float16*       hstate;  // 2 ping-pong buffers of [B, U] (f16)
  float*          cstate;  // [B, U] (f32)
  _Float16*       hseq;    // [S, B, U] output sequence (f16)
};

// ---------------------------------------------------------------------------
// Embedding gather + f32->f16 cast; also builds the time-reversed copy for bw.
// grid (B, S), block E
// ---------------------------------------------------------------------------
__global__ void embed_kernel(const int* __restrict__ tokens,
                             const float* __restrict__ emb,
                             _Float16* __restrict__ x_fw,
                             _Float16* __restrict__ x_bw) {
  const int e = threadIdx.x;
  const int b = blockIdx.x;
  const int s = blockIdx.y;
  const int tf = tokens[b * Ss + s];
  const int tb = tokens[b * Ss + (Ss - 1 - s)];
  x_fw[((size_t)s * Bb + b) * Ee + e] = (_Float16)emb[(size_t)tf * Ee + e];
  x_bw[((size_t)s * Bb + b) * Ee + e] = (_Float16)emb[(size_t)tb * Ee + e];
}

// ---------------------------------------------------------------------------
// Pack [Wk; Wr] (f32, row-major [Kin+U, 4U]) into the WMMA B-operand lane
// layout (f16): for tile (g,nt,kt), lane l holds column g*U+nt*16+(l&15),
// K rows kt*32 + (l>>4)*16 .. +15.  One wave per tile.
// ---------------------------------------------------------------------------
__global__ void pack_w_kernel(const float* __restrict__ Wk,
                              const float* __restrict__ Wr,
                              int Kin, _Float16* __restrict__ dst) {
  const int tid  = blockIdx.x * blockDim.x + threadIdx.x;
  const int lane = tid & 31;
  const int idx  = tid >> 5;
  const int KT   = (Kin + Uu) / 32;
  if (idx >= 4 * NT * KT) return;
  const int kt = idx % KT;
  const int nt = (idx / KT) % NT;
  const int g  = idx / (KT * NT);
  const int col = g * Uu + nt * 16 + (lane & 15);
  const int kb  = kt * 32 + (lane >> 4) * 16;
  v16h v;
#pragma unroll
  for (int j = 0; j < 16; ++j) {
    const int k = kb + j;
    const float w = (k < Kin) ? Wk[(size_t)k * (4 * Uu) + col]
                              : Wr[(size_t)(k - Kin) * (4 * Uu) + col];
    v[j] = (_Float16)w;
  }
  *(v16h*)(dst + ((size_t)idx * 32 + lane) * 16) = v;
}

// ---------------------------------------------------------------------------
// One pipelined LSTM step: blockIdx.y selects {layer0-fw, layer0-bw,
// layer1-fw, layer1-bw}; layer1 runs timestep t-1 (its input h0_seq[t-1] was
// produced by the previous launch), halving the sequential launch count.
//
// Block = 8 waves; wave w owns batch tile mt=w. The nt weight slice (4 gate
// blocks) is staged global->LDS once per step with async-to-LDS; WMMAs read B
// from LDS. The per-wave A fragment stream is split into the x-part and the
// h-part (no per-iteration source select) and software-pipelined at depth 2
// so the global-load latency hides under ~8 WMMAs + LDS reads.
// grid (NT, 4), block 256, dynamic LDS = 4*KT_max*32*16*2 bytes.
// ---------------------------------------------------------------------------
__global__ __launch_bounds__(256) void lstm_step_kernel(DirArgs a0f, DirArgs a0b,
                                                        DirArgs a1f, DirArgs a1b,
                                                        int t) {
  extern __shared__ _Float16 lb[];   // [4][KT][32][16] staged B fragments

  const int role  = blockIdx.y;      // 0:l0 fw, 1:l0 bw, 2:l1 fw, 3:l1 bw
  const int layer = role >> 1;
  const int te    = layer ? (t - 1) : t;
  if (te < 0 || te >= Ss) return;
  const DirArgs d = (role == 0) ? a0f : (role == 1) ? a0b : (role == 2) ? a1f : a1b;
  const int Kin = layer ? Uu : Ee;
  const int KT  = (Kin + Uu) / 32;

  const int tid  = threadIdx.x;
  const int lane = tid & 31;
  const int wave = tid >> 5;         // = mt
  const int nt   = blockIdx.x;

  // ---- stage this nt's weight slice (4 gates x KT k-tiles) into LDS ----
#pragma unroll
  for (int g = 0; g < 4; ++g) {
    const _Float16* gsrc = d.pw + (size_t)(g * NT + nt) * KT * 512;  // KT*32*16 halves
    _Float16* ldst = lb + (size_t)g * KT * 512;
    for (int c = tid; c < KT * 64; c += 256) {                       // 16B chunks
#if USE_ASYNC_LDS
      __builtin_amdgcn_global_load_async_to_lds_b128(
          (gas_v4i*)(uintptr_t)(gsrc + (size_t)c * 8),
          (las_v4i*)(uintptr_t)(ldst + (size_t)c * 8),
          0, 0);
#else
      *(v8h*)(ldst + (size_t)c * 8) = *(const v8h*)(gsrc + (size_t)c * 8);
#endif
    }
  }
#if USE_ASYNC_LDS
#if __has_builtin(__builtin_amdgcn_s_wait_asynccnt)
  __builtin_amdgcn_s_wait_asynccnt(0);
#else
  asm volatile("s_wait_asynccnt 0x0" ::: "memory");
#endif
#endif
  __syncthreads();

  // ---- WMMA main loop ----
  // A lane mapping: row = lane&15; lanes 0-15 hold K {0..15},
  // lanes 16-31 hold K {16..31} of each 32-wide K tile.
  const int row  = wave * 16 + (lane & 15);
  const int koff = (lane >> 4) * 16;
  const _Float16* xrow = d.xseq + ((size_t)te * Bb + row) * Kin;
  const _Float16* hrow = d.hstate + (size_t)(te & 1) * Bb * Uu + (size_t)row * Uu;

  v8f acc0 = {}, acc1 = {}, acc2 = {}, acc3 = {};
  const size_t gs = (size_t)KT * 512;                        // gate stride in LDS (halves)

  // 4 WMMAs against the staged B fragments of k-tile `kt`
  auto do_tile = [&](v16h a, int kt) {
    const _Float16* pb = lb + ((size_t)kt * 32 + lane) * 16;   // gate 0
    const v16h b0 = *(const v16h*)(pb);
    const v16h b1 = *(const v16h*)(pb + gs);
    const v16h b2 = *(const v16h*)(pb + 2 * gs);
    const v16h b3 = *(const v16h*)(pb + 3 * gs);
    acc0 = __builtin_amdgcn_wmma_f32_16x16x32_f16(false, a, false, b0, (short)0, acc0, false, false);
    acc1 = __builtin_amdgcn_wmma_f32_16x16x32_f16(false, a, false, b1, (short)0, acc1, false, false);
    acc2 = __builtin_amdgcn_wmma_f32_16x16x32_f16(false, a, false, b2, (short)0, acc2, false, false);
    acc3 = __builtin_amdgcn_wmma_f32_16x16x32_f16(false, a, false, b3, (short)0, acc3, false, false);
  };

  // Run `KTpart` (even) k-tiles whose A rows come from contiguous `arow`,
  // with a depth-2 software pipeline on the A global loads.
  auto run_part = [&](const _Float16* arow, int ktBase, int KTpart) {
    const _Float16* ap0 = arow + koff;
    const _Float16* ap1 = arow + 32 + koff;
    v8h l0 = *(const v8h*)(ap0), h0 = *(const v8h*)(ap0 + 8);
    v8h l1 = *(const v8h*)(ap1), h1 = *(const v8h*)(ap1 + 8);
    for (int k = 0; k < KTpart; k += 2) {
      // even tile: consume buffer 0, refill with tile k+2
      v16h a;
#pragma unroll
      for (int j = 0; j < 8; ++j) { a[j] = l0[j]; a[8 + j] = h0[j]; }
      if (k + 2 < KTpart) {
        const _Float16* an = arow + (size_t)(k + 2) * 32 + koff;
        l0 = *(const v8h*)(an); h0 = *(const v8h*)(an + 8);
      }
      do_tile(a, ktBase + k);
      // odd tile: consume buffer 1, refill with tile k+3
      v16h a2;
#pragma unroll
      for (int j = 0; j < 8; ++j) { a2[j] = l1[j]; a2[8 + j] = h1[j]; }
      if (k + 3 < KTpart) {
        const _Float16* an = arow + (size_t)(k + 3) * 32 + koff;
        l1 = *(const v8h*)(an); h1 = *(const v8h*)(an + 8);
      }
      do_tile(a2, ktBase + k + 1);
    }
  };

  run_part(xrow, 0, Kin / 32);          // x_t @ Wk part (4 or 16 tiles)
  run_part(hrow, Kin / 32, Uu / 32);    // h_{t-1} @ Wr part (16 tiles)

  // ---- fused LSTM cell update ----
  // C/D layout: lanes 0-15 -> N=lane, M=r; lanes 16-31 -> N=lane-16, M=r+8.
  const int col   = nt * 16 + (lane & 15);
  const int rbase = (lane >= 16) ? 8 : 0;
  _Float16* hout    = d.hstate + (size_t)((te + 1) & 1) * Bb * Uu;
  _Float16* hseqout = d.hseq + (size_t)te * Bb * Uu;

#pragma unroll
  for (int r = 0; r < 8; ++r) {
    const int rrow = wave * 16 + rbase + r;
    const size_t sidx = (size_t)rrow * Uu + col;
    const float zi = acc0[r] + d.bias[0 * Uu + col];
    const float zf = acc1[r] + d.bias[1 * Uu + col];
    const float zg = acc2[r] + d.bias[2 * Uu + col];
    const float zo = acc3[r] + d.bias[3 * Uu + col];
    const float cold = d.cstate[sidx];
    const float ig = 1.0f / (1.0f + __expf(-zi));
    const float fg = 1.0f / (1.0f + __expf(-zf));
    const float og = 1.0f / (1.0f + __expf(-zo));
    const float cn = fg * cold + ig * tanhf(zg);
    d.cstate[sidx] = cn;
    const _Float16 hh = (_Float16)(og * tanhf(cn));
    hout[sidx]    = hh;
    hseqout[sidx] = hh;
  }
}

// ---------------------------------------------------------------------------
// Final dense (1024 -> 3) + softmax. One wave per (b, s) position.
// ---------------------------------------------------------------------------
__global__ void dense_softmax_kernel(const _Float16* __restrict__ h1f,
                                     const _Float16* __restrict__ h1b,
                                     const float* __restrict__ W,
                                     const float* __restrict__ bias,
                                     float* __restrict__ out) {
  const int gid  = blockIdx.x * blockDim.x + threadIdx.x;
  const int wave = gid >> 5;
  const int lane = gid & 31;
  if (wave >= Bb * Ss) return;
  const int b = wave / Ss;
  const int s = wave % Ss;
  const _Float16* hf = h1f + ((size_t)s * Bb + b) * Uu;
  const _Float16* hb = h1b + ((size_t)(Ss - 1 - s) * Bb + b) * Uu;  // bw stored time-reversed
  float a0 = 0.f, a1 = 0.f, a2 = 0.f;
  for (int u = lane; u < Uu; u += 32) {
    const float hv = (float)hf[u];
    a0 += hv * W[(size_t)u * Cc + 0];
    a1 += hv * W[(size_t)u * Cc + 1];
    a2 += hv * W[(size_t)u * Cc + 2];
    const float hbv = (float)hb[u];
    a0 += hbv * W[(size_t)(Uu + u) * Cc + 0];
    a1 += hbv * W[(size_t)(Uu + u) * Cc + 1];
    a2 += hbv * W[(size_t)(Uu + u) * Cc + 2];
  }
#pragma unroll
  for (int m = 16; m >= 1; m >>= 1) {
    a0 += __shfl_xor(a0, m, 32);
    a1 += __shfl_xor(a1, m, 32);
    a2 += __shfl_xor(a2, m, 32);
  }
  if (lane == 0) {
    const float l0 = a0 + bias[0], l1 = a1 + bias[1], l2 = a2 + bias[2];
    const float mx = fmaxf(l0, fmaxf(l1, l2));
    const float e0 = __expf(l0 - mx), e1 = __expf(l1 - mx), e2 = __expf(l2 - mx);
    const float inv = 1.0f / (e0 + e1 + e2);
    float* o = out + (size_t)(b * Ss + s) * Cc;
    o[0] = e0 * inv; o[1] = e1 * inv; o[2] = e2 * inv;
  }
}

// ---------------------------------------------------------------------------
extern "C" void kernel_launch(void* const* d_in, const int* in_sizes, int n_in,
                              void* d_out, int out_size, void* d_ws, size_t ws_size,
                              hipStream_t stream) {
  (void)in_sizes; (void)n_in; (void)out_size; (void)ws_size;
  const int*   tokens  = (const int*)d_in[0];
  const float* emb     = (const float*)d_in[1];
  const float* fw0_Wk  = (const float*)d_in[2];
  const float* fw0_Wr  = (const float*)d_in[3];
  const float* fw0_b   = (const float*)d_in[4];
  const float* fw1_Wk  = (const float*)d_in[5];
  const float* fw1_Wr  = (const float*)d_in[6];
  const float* fw1_b   = (const float*)d_in[7];
  const float* bw0_Wk  = (const float*)d_in[8];
  const float* bw0_Wr  = (const float*)d_in[9];
  const float* bw0_b   = (const float*)d_in[10];
  const float* bw1_Wk  = (const float*)d_in[11];
  const float* bw1_Wr  = (const float*)d_in[12];
  const float* bw1_b   = (const float*)d_in[13];
  const float* dense_W = (const float*)d_in[14];
  const float* dense_b = (const float*)d_in[15];
  float* out = (float*)d_out;

  // --- workspace carve-up (all sizes multiples of 256B) ---
  char* ws = (char*)d_ws;
  size_t off = 0;
  auto alloc = [&](size_t bytes) -> char* {
    char* p = ws + off;
    off += (bytes + 255) & ~(size_t)255;
    return p;
  };
  _Float16* x_fw  = (_Float16*)alloc((size_t)Ss * Bb * Ee * sizeof(_Float16));
  _Float16* x_bw  = (_Float16*)alloc((size_t)Ss * Bb * Ee * sizeof(_Float16));
  _Float16* h0_fw = (_Float16*)alloc((size_t)Ss * Bb * Uu * sizeof(_Float16));
  _Float16* h0_bw = (_Float16*)alloc((size_t)Ss * Bb * Uu * sizeof(_Float16));
  _Float16* h1_fw = (_Float16*)alloc((size_t)Ss * Bb * Uu * sizeof(_Float16));
  _Float16* h1_bw = (_Float16*)alloc((size_t)Ss * Bb * Uu * sizeof(_Float16));

  char* stateBase = ws + off;
  _Float16* hs0f = (_Float16*)alloc(2 * (size_t)Bb * Uu * sizeof(_Float16));
  _Float16* hs0b = (_Float16*)alloc(2 * (size_t)Bb * Uu * sizeof(_Float16));
  _Float16* hs1f = (_Float16*)alloc(2 * (size_t)Bb * Uu * sizeof(_Float16));
  _Float16* hs1b = (_Float16*)alloc(2 * (size_t)Bb * Uu * sizeof(_Float16));
  float* cs0f = (float*)alloc((size_t)Bb * Uu * sizeof(float));
  float* cs0b = (float*)alloc((size_t)Bb * Uu * sizeof(float));
  float* cs1f = (float*)alloc((size_t)Bb * Uu * sizeof(float));
  float* cs1b = (float*)alloc((size_t)Bb * Uu * sizeof(float));
  const size_t stateBytes = (size_t)((ws + off) - stateBase);

  const int KT0 = (Ee + Uu) / 32;   // 20
  const int KT1 = (Uu + Uu) / 32;   // 32
  const size_t pw0Bytes = (size_t)4 * NT * KT0 * 32 * 16 * sizeof(_Float16);
  const size_t pw1Bytes = (size_t)4 * NT * KT1 * 32 * 16 * sizeof(_Float16);
  _Float16* pw_fw0 = (_Float16*)alloc(pw0Bytes);
  _Float16* pw_bw0 = (_Float16*)alloc(pw0Bytes);
  _Float16* pw_fw1 = (_Float16*)alloc(pw1Bytes);
  _Float16* pw_bw1 = (_Float16*)alloc(pw1Bytes);

  // --- zero recurrent state (graph-capturable async memset) ---
  (void)hipMemsetAsync(stateBase, 0, stateBytes, stream);

  // --- embedding gather + cast ---
  embed_kernel<<<dim3(Bb, Ss), Ee, 0, stream>>>(tokens, emb, x_fw, x_bw);

  // --- weight packing into WMMA lane layout ---
  {
    const int waves0 = 4 * NT * KT0;
    const int grid0  = (waves0 * 32 + 255) / 256;
    pack_w_kernel<<<grid0, 256, 0, stream>>>(fw0_Wk, fw0_Wr, Ee, pw_fw0);
    pack_w_kernel<<<grid0, 256, 0, stream>>>(bw0_Wk, bw0_Wr, Ee, pw_bw0);
    const int waves1 = 4 * NT * KT1;
    const int grid1  = (waves1 * 32 + 255) / 256;
    pack_w_kernel<<<grid1, 256, 0, stream>>>(fw1_Wk, fw1_Wr, Uu, pw_fw1);
    pack_w_kernel<<<grid1, 256, 0, stream>>>(bw1_Wk, bw1_Wr, Uu, pw_bw1);
  }

  // --- recurrence: layer0@t and layer1@t-1 fused per launch, both dirs ---
  DirArgs f0{ x_fw,  pw_fw0, fw0_b, hs0f, cs0f, h0_fw };
  DirArgs b0{ x_bw,  pw_bw0, bw0_b, hs0b, cs0b, h0_bw };
  DirArgs f1{ h0_fw, pw_fw1, fw1_b, hs1f, cs1f, h1_fw };
  DirArgs b1{ h0_bw, pw_bw1, bw1_b, hs1b, cs1b, h1_bw };
  const size_t ldsBytes = (size_t)4 * KT1 * 32 * 16 * sizeof(_Float16);  // 128 KB
  for (int t = 0; t <= Ss; ++t)
    lstm_step_kernel<<<dim3(NT, 4), 256, ldsBytes, stream>>>(f0, b0, f1, b1, t);

  // --- dense + softmax ---
  {
    const int threads = Bb * Ss * 32;
    dense_softmax_kernel<<<(threads + 255) / 256, 256, 0, stream>>>(
        h1_fw, h1_bw, dense_W, dense_b, out);
  }
}
